// VectorQuantizer_21139829031574
// MI455X (gfx1250) — matrix operand order repaired and checked by reading, not compiled, not run on hardware
//
#include <hip/hip_runtime.h>
#include <hip/hip_bf16.h>

typedef __attribute__((ext_vector_type(16))) __bf16 v16bf;
typedef __attribute__((ext_vector_type(8)))  float  v8f;

#define WIDTH   64
#define NBINS   2048
#define NROWS   32768            // 16 * 2048 flat_x rows
#define TPB1    512              // 16 waves per block
#define CHUNK   512              // codebook entries resident in LDS per pass
#define NCHUNK  (NBINS / CHUNK)
#define CBS     80               // padded LDS stride (bf16 elems) per codebook entry (160B, 32B-aligned, bank-spread)

// ---------------------------------------------------------------------------
// Kernel 1: per-row argmin over codebook via compensated-bf16 WMMA GEMM.
// Each wave owns one 16-row A tile; codebook chunks are staged in LDS as
// (hi, lo) bf16 pairs + fp32 norms.
// Inner loop minimizes (||c||^2 - 2*dot); the row norm is a per-row constant,
// irrelevant to the argmin and numerically harmful in the comparison, and is
// added back once at the end. The min/argmin epilogue is software-pipelined
// one tile behind the WMMA chain so it co-executes with the matrix ops
// instead of stalling on the WMMA->VALU hazard.
// ---------------------------------------------------------------------------
__global__ __launch_bounds__(TPB1)
void vq_argmin_kernel(const float* __restrict__ x,
                      const float* __restrict__ cb,
                      int*   __restrict__ outIdx,
                      float* __restrict__ outMinD)
{
    extern __shared__ char smem[];
    __bf16* cbh = (__bf16*)smem;                                  // [CHUNK*CBS]
    __bf16* cbl = (__bf16*)(smem + (size_t)CHUNK * CBS * 2);      // [CHUNK*CBS]
    float*  cb2 = (float*)(smem + (size_t)CHUNK * CBS * 4);       // [CHUNK]

    const int tid  = threadIdx.x;
    const int lane = tid & 31;
    const int wave = tid >> 5;
    const int m    = lane & 15;       // A row within tile / B column within tile
    const int h    = lane >> 4;       // half-wave selector

    // ---- A tile: row tile rt covers flat rows [rt*16, rt*16+16) ------------
    const int rt  = blockIdx.x * (TPB1 / 32) + wave;   // 0..2047
    const int row = rt * 16 + m;
    const int nb  = row >> 11;                         // batch index
    const int tt  = row & 2047;                        // time index
    const float* xr = x + (size_t)nb * WIDTH * 2048 + tt;

    // 16-bit A layout (16x32): lane m holds row m; element e -> K=16h+e
    // (contiguous order, matching the B operand).
    v16bf a0h, a0l, a1h, a1l;
    float ss = 0.f;
    #pragma unroll
    for (int e = 0; e < 16; ++e) {
        const int k0 = 16 * h + e;                         // K chunk 0 (K<32)
        const float v0 = xr[(size_t)k0 * 2048];
        const float v1 = xr[(size_t)(k0 + 32) * 2048];     // K chunk 1
        ss += v0 * v0 + v1 * v1;
        const __bf16 h0 = (__bf16)v0;
        const __bf16 h1 = (__bf16)v1;
        a0h[e] = h0; a0l[e] = (__bf16)(v0 - (float)h0);
        a1h[e] = h1; a1l[e] = (__bf16)(v1 - (float)h1);
    }
    ss += __shfl_xor(ss, 16, 32);      // full fp32 ||x_row||^2 (row = m)
    // redistribute norms to match C layout: VGPR i of this lane is row 8h+i
    float n2[8];
    #pragma unroll
    for (int i = 0; i < 8; ++i) n2[i] = __shfl(ss, 8 * h + i, 32);

    float bestD[8];                   // tracks (||c||^2 - 2*dot), NOT full d2
    int   bestI[8];
    #pragma unroll
    for (int i = 0; i < 8; ++i) { bestD[i] = 3.0e38f; bestI[i] = 0; }

    union B32 { uint4 u[2]; v16bf v; };

    // software pipeline state: accumulator/metadata of the previous tile.
    // c2P = +3e38 makes the priming epilogue a no-op under strict '<'.
    v8f   cP   = {};
    float c2P  = 3.0e38f;
    int   colP = 0;

    for (int cc = 0; cc < NCHUNK; ++cc) {
        __syncthreads();
        // stage codebook chunk: fp32 -> (hi, lo) bf16 + exact fp32 norm
        for (int e = tid; e < CHUNK; e += TPB1) {
            const float* cr = cb + (size_t)(cc * CHUNK + e) * WIDTH;
            float s2 = 0.f;
            #pragma unroll 4
            for (int k = 0; k < WIDTH; ++k) {
                const float v = cr[k];
                s2 += v * v;
                const __bf16 hi = (__bf16)v;
                cbh[e * CBS + k] = hi;
                cbl[e * CBS + k] = (__bf16)(v - (float)hi);
            }
            cb2[e] = s2;
        }
        __syncthreads();

        // prefetch next chunk toward L2/L0 while we compute on this one
        if (cc + 1 < NCHUNK) {
            const float* nxt = cb + (size_t)(cc + 1) * CHUNK * WIDTH;
            __builtin_prefetch(nxt + (size_t)tid * 64, 0, 1);   // 512 threads x 256B = 128KB chunk
        }

        for (int nt = 0; nt < CHUNK / 16; ++nt) {
            const int colbase = nt * 16;
            // B layout (32x16): lane holds column N=m; lanes 0-15 K=0..15,
            // lanes 16-31 K=16..31 -> one contiguous 32B run per chunk.
            const __bf16* ph = cbh + (colbase + m) * CBS + 16 * h;
            const __bf16* pl = cbl + (colbase + m) * CBS + 16 * h;
            B32 b0h, b1h, b0l, b1l;
            b0h.u[0] = ((const uint4*)ph)[0];        b0h.u[1] = ((const uint4*)ph)[1];
            b1h.u[0] = ((const uint4*)(ph + 32))[0]; b1h.u[1] = ((const uint4*)(ph + 32))[1];
            b0l.u[0] = ((const uint4*)pl)[0];        b0l.u[1] = ((const uint4*)pl)[1];
            b1l.u[0] = ((const uint4*)(pl + 32))[0]; b1l.u[1] = ((const uint4*)(pl + 32))[1];
            const int   col = cc * CHUNK + colbase + m;
            const float c2  = cb2[colbase + m];

            v8f c = {};
            c = __builtin_amdgcn_wmma_f32_16x16x32_bf16(false, a0h, false, b0h.v, (short)0, c, false, false);
            c = __builtin_amdgcn_wmma_f32_16x16x32_bf16(false, a1h, false, b1h.v, (short)0, c, false, false);
            c = __builtin_amdgcn_wmma_f32_16x16x32_bf16(false, a0h, false, b0l.v, (short)0, c, false, false);
            c = __builtin_amdgcn_wmma_f32_16x16x32_bf16(false, a1h, false, b1l.v, (short)0, c, false, false);
            c = __builtin_amdgcn_wmma_f32_16x16x32_bf16(false, a0l, false, b0h.v, (short)0, c, false, false);
            c = __builtin_amdgcn_wmma_f32_16x16x32_bf16(false, a1l, false, b1h.v, (short)0, c, false, false);

            // epilogue of the PREVIOUS tile: independent of the in-flight
            // WMMA chain above, so it fills the hazard/co-exec slots.
            #pragma unroll
            for (int i = 0; i < 8; ++i) {
                const float dd = __builtin_fmaf(-2.f, cP[i], c2P);   // c2 - 2*dot
                if (dd < bestD[i]) { bestD[i] = dd; bestI[i] = colP; }
            }

            cP = c; c2P = c2; colP = col;
        }
    }

    // drain: epilogue of the final tile
    #pragma unroll
    for (int i = 0; i < 8; ++i) {
        const float dd = __builtin_fmaf(-2.f, cP[i], c2P);
        if (dd < bestD[i]) { bestD[i] = dd; bestI[i] = colP; }
    }

    // reduce (min, argmin-first) across the 16 lanes of each half-wave
    #pragma unroll
    for (int i = 0; i < 8; ++i) {
        float d  = n2[i] + bestD[i];     // restore full d2 = ||x||^2 + (c2 - 2*dot)
        int   bi = bestI[i];
        #pragma unroll
        for (int off = 1; off < 16; off <<= 1) {
            const float od = __shfl_xor(d, off, 32);
            const int   oi = __shfl_xor(bi, off, 32);
            if (od < d || (od == d && oi < bi)) { d = od; bi = oi; }
        }
        if (m == 0) {
            const int r = rt * 16 + 8 * h + i;   // lane0 -> rows 0..7, lane16 -> rows 8..15
            outIdx[r]  = bi;
            outMinD[r] = sqrtf(fmaxf(d, 0.f));
        }
    }
}

// ---------------------------------------------------------------------------
// Kernel 2: gather quantized output (raw-view reshape: out[i] = cb[idx[i/64]][i%64]),
// straight-through add, and deterministic per-block MSE partial sums.
// ---------------------------------------------------------------------------
__global__ __launch_bounds__(256)
void vq_quant_kernel(const float* __restrict__ x,
                     const float* __restrict__ cb,
                     const int*   __restrict__ idx,
                     float* __restrict__ out,
                     float* __restrict__ partials)
{
    __shared__ float red[256];
    const int tid = threadIdx.x;
    const size_t i0 = (size_t)blockIdx.x * 2048 + (size_t)tid * 8;   // 8 elems, same idx row
    const int r  = idx[i0 >> 6];
    const int c0 = (int)(i0 & 63);
    const float4* qp = (const float4*)(cb + (size_t)r * WIDTH + c0);
    const float4* xp = (const float4*)(x + i0);
    const float4 q0 = qp[0], q1 = qp[1];
    const float4 x0 = xp[0], x1 = xp[1];
    const float qv[8] = {q0.x, q0.y, q0.z, q0.w, q1.x, q1.y, q1.z, q1.w};
    const float xv[8] = {x0.x, x0.y, x0.z, x0.w, x1.x, x1.y, x1.z, x1.w};
    float s = 0.f, o[8];
    #pragma unroll
    for (int e = 0; e < 8; ++e) {
        const float d = qv[e] - xv[e];
        s += d * d;
        o[e] = xv[e] + d;     // x + (quantized - x), bit-faithful to reference
    }
    float4* op = (float4*)(out + i0);
    op[0] = make_float4(o[0], o[1], o[2], o[3]);
    op[1] = make_float4(o[4], o[5], o[6], o[7]);

    red[tid] = s;
    __syncthreads();
    for (int st = 128; st > 0; st >>= 1) {
        if (tid < st) red[tid] += red[tid + st];
        __syncthreads();
    }
    if (tid == 0) partials[blockIdx.x] = red[0];
}

// ---------------------------------------------------------------------------
// Kernel 3: deterministic final reduction -> loss, fit scalars.
// ---------------------------------------------------------------------------
__global__ __launch_bounds__(256)
void vq_finalize_kernel(const float* __restrict__ partials,
                        const float* __restrict__ mind,
                        float* __restrict__ outScalars)
{
    __shared__ float redL[256];
    __shared__ float redF[256];
    const int tid = threadIdx.x;
    float sl = 0.f;
    for (int j = tid; j < 1024; j += 256) sl += partials[j];
    float sf = 0.f;
    for (int j = tid; j < NROWS; j += 256) sf += mind[j];
    redL[tid] = sl; redF[tid] = sf;
    __syncthreads();
    for (int st = 128; st > 0; st >>= 1) {
        if (tid < st) { redL[tid] += redL[tid + st]; redF[tid] += redF[tid + st]; }
        __syncthreads();
    }
    if (tid == 0) {
        outScalars[0] = 1.25f * redL[0] / 2097152.0f;  // codebook_loss + 0.25*commit_loss
        outScalars[1] = redF[0] / (float)NROWS;        // fit = mean min distance
    }
}

// ---------------------------------------------------------------------------
extern "C" void kernel_launch(void* const* d_in, const int* in_sizes, int n_in,
                              void* d_out, int out_size, void* d_ws, size_t ws_size,
                              hipStream_t stream)
{
    const float* x  = (const float*)d_in[0];   // (16, 64, 2048) fp32
    const float* cb = (const float*)d_in[1];   // (2048, 64) fp32
    float* out = (float*)d_out;                // [2097152 quantized][loss][fit]

    int*   widx  = (int*)d_ws;                            // 32768 ints
    float* wmind = (float*)((char*)d_ws + 131072);        // 32768 floats
    float* wpart = (float*)((char*)d_ws + 262144);        // 1024 floats

    const size_t lds = (size_t)CHUNK * CBS * 2 * 2 + (size_t)CHUNK * 4;  // hi+lo bf16 + fp32 norms

    vq_argmin_kernel<<<dim3(128), dim3(TPB1), lds, stream>>>(x, cb, widx, wmind);
    vq_quant_kernel<<<dim3(1024), dim3(256), 0, stream>>>(x, cb, widx, out, wpart);
    vq_finalize_kernel<<<dim3(1), dim3(256), 0, stream>>>(wpart, wmind, out + 2097152);
}
